// GPT_X_48773648613997
// MI455X (gfx1250) — matrix operand order repaired and checked by reading, not compiled, not run on hardware
//
#include <hip/hip_runtime.h>
#include <hip/hip_bf16.h>
#include <math.h>

// ---------------------------------------------------------------------------
// GPT-2 small forward + loss for MI455X (gfx1250, wave32, WMMA).
// All GEMMs run through v_wmma_f32_16x16x32_f16 (f16 inputs, f32 accum).
// 256-thread blocks = 8 waves, 128x64 block tile, 32x32 per wave.
// Double-buffered LDS pipeline: next tile staged in registers (b128 loads)
// while current tile feeds the WMMAs; one barrier per K-step.
// ---------------------------------------------------------------------------

typedef __attribute__((ext_vector_type(16))) _Float16 v16h;
typedef __attribute__((ext_vector_type(4)))  _Float16 v4h;
typedef __attribute__((ext_vector_type(8)))  float    v8f;

#define GPT_V  50257
#define GPT_S  1024
#define GPT_L  12
#define GPT_H  12
#define GPT_E  768
#define GPT_B  2
#define GPT_FF 3072
#define GPT_D  64
#define GPT_BS (GPT_B * GPT_S)   // 2048 rows

#define TILE_M 128
#define TILE_N 64
#define TILE_K 32
#define LDS_PAD 8                 // rows padded to 40 halfs (80 B, 8B aligned)

#define FUSE_GELU 1
#define FUSE_RES  2

// ---------------------------- reductions -----------------------------------
__device__ __forceinline__ float wave_sum32(float v) {
#pragma unroll
  for (int o = 16; o > 0; o >>= 1) v += __shfl_xor(v, o, 32);
  return v;
}
__device__ __forceinline__ float wave_max32(float v) {
#pragma unroll
  for (int o = 16; o > 0; o >>= 1) v = fmaxf(v, __shfl_xor(v, o, 32));
  return v;
}
__device__ float block_sum256(float v, float* red) {
  v = wave_sum32(v);
  int w = threadIdx.x >> 5, l = threadIdx.x & 31;
  __syncthreads();
  if (l == 0) red[w] = v;
  __syncthreads();
  if (threadIdx.x == 0) {
    float t = 0.f;
#pragma unroll
    for (int i = 0; i < 8; ++i) t += red[i];
    red[0] = t;
  }
  __syncthreads();
  return red[0];
}
__device__ float block_max256(float v, float* red) {
  v = wave_max32(v);
  int w = threadIdx.x >> 5, l = threadIdx.x & 31;
  __syncthreads();
  if (l == 0) red[w] = v;
  __syncthreads();
  if (threadIdx.x == 0) {
    float t = -INFINITY;
#pragma unroll
    for (int i = 0; i < 8; ++i) t = fmaxf(t, red[i]);
    red[0] = t;
  }
  __syncthreads();
  return red[0];
}

// ---------------------------- embedding ------------------------------------
__global__ __launch_bounds__(256) void embed_kernel(
    const int* __restrict__ ids, const float* __restrict__ wte,
    const float* __restrict__ wpe, float* __restrict__ x) {
  int row = blockIdx.x;                  // b*S + s
  int s   = row % GPT_S;
  int tok = ids[row];
  const float* te = wte + (size_t)tok * GPT_E;
  const float* pe = wpe + (size_t)s * GPT_E;
  float* xr = x + (size_t)row * GPT_E;
  for (int i = threadIdx.x; i < GPT_E; i += 256) xr[i] = te[i] + pe[i];
}

// ---------------------------- layernorm ------------------------------------
__global__ __launch_bounds__(256) void layernorm_kernel(
    const float* __restrict__ x, const float* __restrict__ w,
    const float* __restrict__ b, float* __restrict__ y) {
  __shared__ float red[8];
  int row = blockIdx.x;
  const float* xr = x + (size_t)row * GPT_E;
  float* yr = y + (size_t)row * GPT_E;
  float s = 0.f;
  for (int i = threadIdx.x; i < GPT_E; i += 256) s += xr[i];
  float mu = block_sum256(s, red) * (1.f / GPT_E);
  float v = 0.f;
  for (int i = threadIdx.x; i < GPT_E; i += 256) {
    float d = xr[i] - mu;
    v += d * d;
  }
  float var = block_sum256(v, red) * (1.f / GPT_E);
  float inv = rsqrtf(var + 1e-5f);
  for (int i = threadIdx.x; i < GPT_E; i += 256)
    yr[i] = (xr[i] - mu) * inv * w[i] + b[i];
}

// ---------------------------- WMMA GEMM ------------------------------------
// C[M,N] = A[M,K] @ W[N,K]^T (+bias) (+residual) (+gelu); f16 in, f32 accum.
// Requires M % 128 == 0, K % 32 == 0. N boundary handled (zero-fill / guard).
__global__ __launch_bounds__(256) void gemm_wmma_kernel(
    const float* __restrict__ A, const float* __restrict__ W,
    const float* __restrict__ bias, const float* __restrict__ residual,
    float* __restrict__ C, int M, int N, int K, int fuse) {
  __shared__ _Float16 As[2][TILE_M][TILE_K + LDS_PAD];   // [buf][m][k]
  __shared__ _Float16 Bs[2][TILE_N][TILE_K + LDS_PAD];   // [buf][n][k]

  const int tid  = threadIdx.x;
  const int wave = tid >> 5;           // 0..7
  const int lane = tid & 31;
  const int wm   = (wave & 3) * 32;    // wave's m offset inside 128x64 tile
  const int wn   = (wave >> 2) * 32;   // wave's n offset
  const int m0   = blockIdx.y * TILE_M;
  const int n0   = blockIdx.x * TILE_N;

  v8f acc[2][2] = {};                  // 2x2 grid of 16x16 f32 tiles

  const int fm   = lane & 15;                    // fragment row (A) / col (B)
  const int khiA = (lane >> 4) ? 16 : 0;         // A K-phase per half-wave
  const int kB   = (lane >> 4) ? 16 : 0;         // B K-phase per half-wave

  // per-thread staging coordinates (element index = multiple of 4)
  int ar[4], ac[4], br[2], bc[2];
#pragma unroll
  for (int p = 0; p < 4; ++p) {
    int e = (p * 256 + tid) * 4;
    ar[p] = e >> 5;
    ac[p] = e & 31;
  }
#pragma unroll
  for (int p = 0; p < 2; ++p) {
    int e = (p * 256 + tid) * 4;
    br[p] = e >> 5;
    bc[p] = e & 31;
  }

  float4 ra[4], rb[2];
  // batched register-stage of one K-tile (independent b128 loads)
  auto load_tile = [&](int kk) {
#pragma unroll
    for (int p = 0; p < 4; ++p)
      ra[p] = *(const float4*)(A + (size_t)(m0 + ar[p]) * K + kk + ac[p]);
#pragma unroll
    for (int p = 0; p < 2; ++p) {
      int n = n0 + br[p];
      rb[p] = make_float4(0.f, 0.f, 0.f, 0.f);
      if (n < N) rb[p] = *(const float4*)(W + (size_t)n * K + kk + bc[p]);
    }
  };

  const int nk = K / TILE_K;
  load_tile(0);

  for (int kt = 0; kt < nk; ++kt) {
    const int w = kt & 1;
    // ---- commit staged registers to LDS buffer w (cvt_pk + ds_store_b64)
#pragma unroll
    for (int p = 0; p < 4; ++p) {
      v4h h = {(_Float16)ra[p].x, (_Float16)ra[p].y,
               (_Float16)ra[p].z, (_Float16)ra[p].w};
      *(v4h*)&As[w][ar[p]][ac[p]] = h;
    }
#pragma unroll
    for (int p = 0; p < 2; ++p) {
      v4h h = {(_Float16)rb[p].x, (_Float16)rb[p].y,
               (_Float16)rb[p].z, (_Float16)rb[p].w};
      *(v4h*)&Bs[w][br[p]][bc[p]] = h;
    }
    __syncthreads();
    // ---- issue next tile's global loads; waited on only next iteration
    if (kt + 1 < nk) load_tile((kt + 1) * TILE_K);

    // ---- fragments in contiguous k order (k = 16*half + s, A and B alike) + 4 WMMAs from buffer w
    v16h af[2], bf[2];
#pragma unroll
    for (int t = 0; t < 2; ++t) {
      const _Float16* ap = &As[w][wm + t * 16 + fm][0];
#pragma unroll
      for (int p = 0; p < 4; ++p) {            // VGPR 0..3 and 4..7
        af[t][2 * p]     = ap[khiA + 2 * p];
        af[t][2 * p + 1] = ap[khiA + 2 * p + 1];
        af[t][8 + 2 * p]     = ap[khiA + 8 + 2 * p];
        af[t][8 + 2 * p + 1] = ap[khiA + 8 + 2 * p + 1];
      }
      const _Float16* bp = &Bs[w][wn + t * 16 + fm][0];
#pragma unroll
      for (int p = 0; p < 8; ++p) {
        bf[t][2 * p]     = bp[kB + 2 * p];
        bf[t][2 * p + 1] = bp[kB + 2 * p + 1];
      }
    }
#pragma unroll
    for (int i = 0; i < 2; ++i)
#pragma unroll
      for (int j = 0; j < 2; ++j)
        acc[i][j] = __builtin_amdgcn_wmma_f32_16x16x32_f16(
            false, af[i], false, bf[j], (short)0, acc[i][j], false, false);
    // no trailing barrier: buffer w^1 was last read one iteration ago and is
    // only rewritten after the next iteration's barrier
  }

  // ---- epilogue: C/D layout = VGPR r -> M=r (lanes 0..15) / M=r+8 (16..31)
  const float kGelu = 0.7978845608028654f;   // sqrt(2/pi)
  const int nlane = lane & 15;
  const int moff  = (lane >> 4) ? 8 : 0;
#pragma unroll
  for (int i = 0; i < 2; ++i)
#pragma unroll
    for (int j = 0; j < 2; ++j) {
      int n = n0 + wn + j * 16 + nlane;
      if (n >= N) continue;
      int mbase = m0 + wm + i * 16 + moff;
      float bv = bias ? bias[n] : 0.f;
#pragma unroll
      for (int r = 0; r < 8; ++r) {
        int m = mbase + r;
        float v = acc[i][j][r] + bv;
        if (fuse & FUSE_GELU) {
          float u = kGelu * (v + 0.044715f * v * v * v);
          v = 0.5f * v * (1.f + tanhf(u));
        }
        if (fuse & FUSE_RES) v += residual[(size_t)m * N + n];
        C[(size_t)m * N + n] = v;
      }
    }
}

// ---------------------------- attention ------------------------------------
// One wave per query row; online (flash) softmax; D=64 -> 2 floats per lane.
__global__ __launch_bounds__(256) void attention_kernel(
    const float* __restrict__ qkv, float* __restrict__ out) {
  const int wave = threadIdx.x >> 5;
  const int lane = threadIdx.x & 31;
  const long long gq = (long long)blockIdx.x * 8 + wave;  // over B*H*S
  const int q  = (int)(gq % GPT_S);
  const int bh = (int)(gq / GPT_S);
  const int h  = bh % GPT_H;
  const int b  = bh / GPT_H;
  const int ld = 3 * GPT_E;

  const float* qrow  = qkv + (size_t)(b * GPT_S + q) * ld + h * GPT_D;
  const float* kbase = qkv + (size_t)b * GPT_S * ld + GPT_E + h * GPT_D;
  const float* vbase = qkv + (size_t)b * GPT_S * ld + 2 * GPT_E + h * GPT_D;

  const float scale = rsqrtf((float)GPT_D);
  const float q0 = qrow[lane] * scale;
  const float q1 = qrow[lane + 32] * scale;

  float mx = -INFINITY, sum = 0.f, a0 = 0.f, a1 = 0.f;
  for (int j = 0; j <= q; ++j) {
    const float* kr = kbase + (size_t)j * ld;
    float d = q0 * kr[lane] + q1 * kr[lane + 32];
    d = wave_sum32(d);
    float nm   = fmaxf(mx, d);
    float corr = __expf(mx - nm);      // exp(-inf)=0 handles first iteration
    float p    = __expf(d - nm);
    const float* vr = vbase + (size_t)j * ld;
    sum = sum * corr + p;
    a0  = a0 * corr + p * vr[lane];
    a1  = a1 * corr + p * vr[lane + 32];
    mx = nm;
  }
  float inv = 1.f / sum;
  float* o = out + (size_t)(b * GPT_S + q) * GPT_E + h * GPT_D;
  o[lane] = a0 * inv;
  o[lane + 32] = a1 * inv;
}

// ---------------------------- loss -----------------------------------------
__global__ __launch_bounds__(256) void nll_rows_kernel(
    const float* __restrict__ logits, const int* __restrict__ target,
    float* __restrict__ nll, float* __restrict__ vld) {
  __shared__ float red[8];
  int row = blockIdx.x;
  const float* lr = logits + (size_t)row * GPT_V;
  float mx = -INFINITY;
  for (int i = threadIdx.x; i < GPT_V; i += 256) mx = fmaxf(mx, lr[i]);
  mx = block_max256(mx, red);
  float s = 0.f;
  for (int i = threadIdx.x; i < GPT_V; i += 256) s += __expf(lr[i] - mx);
  s = block_sum256(s, red);
  if (threadIdx.x == 0) {
    int t = target[row];
    if (t >= 0) {
      nll[row] = logf(s) + mx - lr[t];
      vld[row] = 1.f;
    } else {
      nll[row] = 0.f;
      vld[row] = 0.f;
    }
  }
}

__global__ __launch_bounds__(256) void loss_reduce_kernel(
    const float* __restrict__ nll, const float* __restrict__ vld,
    float* __restrict__ out) {
  __shared__ float redA[8];
  __shared__ float redB[8];
  float s = 0.f, c = 0.f;
  for (int i = threadIdx.x; i < GPT_BS; i += 256) {
    s += nll[i];
    c += vld[i];
  }
  s = block_sum256(s, redA);
  c = block_sum256(c, redB);
  if (threadIdx.x == 0) out[0] = s / fmaxf(c, 1.f);
}

// ---------------------------- host side ------------------------------------
static inline void launch_gemm(const float* A, const float* W, const float* bias,
                               const float* res, float* C, int M, int N, int K,
                               int fuse, hipStream_t st) {
  dim3 grid((N + TILE_N - 1) / TILE_N, M / TILE_M);
  gemm_wmma_kernel<<<grid, dim3(256), 0, st>>>(A, W, bias, res, C, M, N, K, fuse);
}

extern "C" void kernel_launch(void* const* d_in, const int* in_sizes, int n_in,
                              void* d_out, int out_size, void* d_ws, size_t ws_size,
                              hipStream_t stream) {
  const int*   ids    = (const int*)  d_in[0];
  const int*   target = (const int*)  d_in[1];
  const float* wte    = (const float*)d_in[2];
  const float* wpe    = (const float*)d_in[3];
  const float* ln1_w  = (const float*)d_in[4];
  const float* ln1_b  = (const float*)d_in[5];
  const float* attn_w = (const float*)d_in[6];
  const float* attn_b = (const float*)d_in[7];
  const float* proj_w = (const float*)d_in[8];
  const float* proj_b = (const float*)d_in[9];
  const float* ln2_w  = (const float*)d_in[10];
  const float* ln2_b  = (const float*)d_in[11];
  const float* fc_w   = (const float*)d_in[12];
  const float* fc_b   = (const float*)d_in[13];
  const float* fc2_w  = (const float*)d_in[14];
  const float* fc2_b  = (const float*)d_in[15];
  const float* lnf_w  = (const float*)d_in[16];
  const float* lnf_b  = (const float*)d_in[17];
  const float* lm_w   = (const float*)d_in[18];

  float* logits = (float*)d_out;                       // [BS, V]
  float* loss   = logits + (size_t)GPT_BS * GPT_V;     // scalar

  // workspace layout (floats)
  float* X   = (float*)d_ws;                           // [BS, E]
  float* Hb  = X   + (size_t)GPT_BS * GPT_E;           // [BS, E]
  float* QKV = Hb  + (size_t)GPT_BS * GPT_E;           // [BS, 3E]
  float* ATT = QKV + (size_t)GPT_BS * 3 * GPT_E;       // [BS, E]
  float* FC  = ATT + (size_t)GPT_BS * GPT_E;           // [BS, FF]
  float* NLL = FC  + (size_t)GPT_BS * GPT_FF;          // [BS]
  float* VLD = NLL + GPT_BS;                           // [BS]

  // x = wte[ids] + wpe
  embed_kernel<<<GPT_BS, 256, 0, stream>>>(ids, wte, wpe, X);

  for (int l = 0; l < GPT_L; ++l) {
    const float* aw = attn_w + (size_t)l * 3 * GPT_E * GPT_E;
    const float* ab = attn_b + (size_t)l * 3 * GPT_E;
    const float* pw = proj_w + (size_t)l * GPT_E * GPT_E;
    const float* pb = proj_b + (size_t)l * GPT_E;
    const float* fw = fc_w   + (size_t)l * GPT_FF * GPT_E;
    const float* fb = fc_b   + (size_t)l * GPT_FF;
    const float* gw = fc2_w  + (size_t)l * GPT_E * GPT_FF;
    const float* gb = fc2_b  + (size_t)l * GPT_E;

    // attention block
    layernorm_kernel<<<GPT_BS, 256, 0, stream>>>(X, ln1_w + l * GPT_E,
                                                 ln1_b + l * GPT_E, Hb);
    launch_gemm(Hb, aw, ab, nullptr, QKV, GPT_BS, 3 * GPT_E, GPT_E, 0, stream);
    attention_kernel<<<GPT_B * GPT_H * GPT_S / 8, 256, 0, stream>>>(QKV, ATT);
    launch_gemm(ATT, pw, pb, X, X, GPT_BS, GPT_E, GPT_E, FUSE_RES, stream);

    // MLP block
    layernorm_kernel<<<GPT_BS, 256, 0, stream>>>(X, ln2_w + l * GPT_E,
                                                 ln2_b + l * GPT_E, Hb);
    launch_gemm(Hb, fw, fb, nullptr, FC, GPT_BS, GPT_FF, GPT_E, FUSE_GELU, stream);
    launch_gemm(FC, gw, gb, X, X, GPT_BS, GPT_E, GPT_FF, FUSE_RES, stream);
  }

  // final LN + LM head + loss
  layernorm_kernel<<<GPT_BS, 256, 0, stream>>>(X, lnf_w, lnf_b, Hb);
  launch_gemm(Hb, lm_w, nullptr, nullptr, logits, GPT_BS, GPT_V, GPT_E, 0, stream);
  nll_rows_kernel<<<GPT_BS, 256, 0, stream>>>(logits, target, NLL, VLD);
  loss_reduce_kernel<<<1, 256, 0, stream>>>(NLL, VLD, loss);
}
